// QuantizedLinear_88605175316808
// MI455X (gfx1250) — matrix operand order repaired and checked by reading, not compiled, not run on hardware
//
#include <hip/hip_runtime.h>

// ---------------------------------------------------------------------------
// QuantizedLinear for MI455X (gfx1250, wave32, WMMA + TDM)
//   out[m,n] = scale * sum_k x[m,k] * q[n,k],  q = round(W/scale) in [-127,127]
//   q stored exactly as fp16; x split into hi+lo fp16; two f16 WMMA chains
//   accumulate in f32 (near-fp32 accuracy at fp16-WMMA rate / 2).
//   B tiles DMA'd LDS-ward by the Tensor Data Mover (with TDM padding for a
//   conflict-free LDS stride); X tiles register-double-buffered + converted.
// ---------------------------------------------------------------------------

typedef __attribute__((ext_vector_type(16))) _Float16 v16h;
typedef __attribute__((ext_vector_type(8)))  _Float16 v8h;
typedef __attribute__((ext_vector_type(4)))  _Float16 v4h;
typedef __attribute__((ext_vector_type(8)))  float    v8f;
typedef __attribute__((ext_vector_type(4)))  float    v4f;
typedef __attribute__((ext_vector_type(4)))  unsigned u32x4;
typedef __attribute__((ext_vector_type(8)))  unsigned u32x8;

#define K_DIM 2048
#define N_DIM 2048
#define BM 128
#define BN 128
#define BK 32
#define LDA 40   // 32 halfs + 8 pad = 80B row stride (16B aligned, conflict-free)

// ---------------- Kernel 1: abs-max of weight -> float bits in ws ----------
__global__ __launch_bounds__(256)
void wabsmax_kernel(const float* __restrict__ w, unsigned* __restrict__ absbits, int n) {
    __shared__ float red[256];
    const int t = threadIdx.x;
    float m = 0.0f;
    for (int i = blockIdx.x * 256 + t; i < n; i += gridDim.x * 256)
        m = fmaxf(m, fabsf(w[i]));
    red[t] = m;
    __syncthreads();
    for (int s = 128; s > 0; s >>= 1) {
        if (t < s) red[t] = fmaxf(red[t], red[t + s]);
        __syncthreads();
    }
    if (t == 0) atomicMax(absbits, __float_as_uint(red[0]));  // abs floats: bits are order-preserving
}

// ---------------- Kernel 2: q[n,k] = (fp16) rint(W/scale) ------------------
__global__ __launch_bounds__(256)
void wquant_kernel(const float* __restrict__ w, const unsigned* __restrict__ absbits,
                   _Float16* __restrict__ q, int n) {
    const float amax = __uint_as_float(*absbits);
    const float inv  = (amax > 0.0f) ? (127.0f / amax) : 0.0f;
    const int i = (blockIdx.x * 256 + threadIdx.x) * 4;
    if (i < n) {
        v4f v = *(const v4f*)(w + i);
        v4h o;
        o.x = (_Float16)rintf(v.x * inv);   // rint == round-half-even (matches jnp.round)
        o.y = (_Float16)rintf(v.y * inv);
        o.z = (_Float16)rintf(v.z * inv);
        o.w = (_Float16)rintf(v.w * inv);
        *(v4h*)(q + i) = o;
    }
}

// ---------------- Kernel 3: double-buffered WMMA GEMM with TDM B-path ------
__global__ __launch_bounds__(256)
void qlinear_wmma_kernel(const float* __restrict__ X,
                         const _Float16* __restrict__ Q,
                         const unsigned* __restrict__ absbits,
                         float* __restrict__ C) {
    __shared__ __align__(16) _Float16 sAhi[2][BM][LDA];
    __shared__ __align__(16) _Float16 sAlo[2][BM][LDA];
    __shared__ __align__(16) _Float16 sB [2][BN][LDA];

    const int t    = threadIdx.x;
    const int lane = t & 31;
    const int wave = t >> 5;          // 8 waves
    const int wm   = wave >> 2;       // 0..1  -> 64 rows each
    const int wn   = wave & 3;        // 0..3  -> 32 cols each
    const long blockM = (long)blockIdx.y * BM;
    const long blockN = (long)blockIdx.x * BN;

    const float scale = __uint_as_float(*absbits) * (1.0f / 127.0f);

    v8f acc[4][2];
#pragma unroll
    for (int i = 0; i < 4; ++i)
#pragma unroll
        for (int j = 0; j < 2; ++j)
            acc[i][j] = (v8f){0.f,0.f,0.f,0.f,0.f,0.f,0.f,0.f};

    // cooperative X-load coordinates
    const int arow = t >> 3;          // 0..31 (x4 passes -> 128 rows)
    const int acv  = t & 7;           // float4 column within 32-wide K tile

    // fragment coordinates (documented wave32 WMMA layouts)
    const int amr = wm * 64 + (lane & 15);
    const int kA  = (lane < 16) ? 0 : 8;    // A 16x32 f16: half-wave K interleave
    const int kB  = (lane < 16) ? 0 : 8;    // B 32x16 f16: half-wave K interleave
    const int bnr = wn * 32 + (lane & 15);

    // ---- TDM descriptor issue for a B tile (wave 0 only; EXEC ignored) ----
    auto tdm_load_B = [&](int kb, int buf) {
        if (t < 32) {
            unsigned long long ga =
                (unsigned long long)(const void*)(Q + blockN * K_DIM + kb);
            unsigned lds = (unsigned)(unsigned long long)(const void*)&sB[buf][0][0];
            u32x4 g0;
            g0[0] = 1u;                                   // count=1, user descriptor
            g0[1] = lds;                                  // lds_addr (bytes)
            g0[2] = (unsigned)ga;                         // global_addr[31:0]
            g0[3] = (unsigned)(ga >> 32) | 0x80000000u;   // addr[56:32] | type=2
            u32x8 g1;
            g1[0] = (1u << 16)       // data_size = 2 bytes
                  | (1u << 20)       // pad_enable
                  | (3u << 22)       // pad_interval: 16 DWORDs (= 64B = one 32-half row)
                  | (3u << 25);      // pad_amount: 4 DWORDs (= 16B = 8-half pad -> LDA=40)
            g1[1] = (unsigned)(K_DIM) << 16;   // tensor_dim0[15:0] (bits 63:48)
            g1[2] = (unsigned)(N_DIM) << 16;   // dim0 hi=0 | tensor_dim1[15:0]
            g1[3] = (unsigned)(BK) << 16;      // dim1 hi=0 | tile_dim0 = 32
            g1[4] = (unsigned)(BN);            // tile_dim1 = 128 | tile_dim2 = 0
            g1[5] = (unsigned)(K_DIM);         // tensor_dim0_stride[31:0]
            g1[6] = 0u;                        // stride hi | tensor_dim1_stride lo
            g1[7] = 0u;
            asm volatile("tensor_load_to_lds %0, %1" :: "s"(g0), "s"(g1) : "memory");
        }
    };
    auto tdm_wait = [&]() {
        if (t < 32) __builtin_amdgcn_s_wait_tensorcnt(0);
    };

    // ---- X tile: global -> regs, then convert/split -> LDS ----
    v4f xv[4];
    auto load_X = [&](int kb) {
#pragma unroll
        for (int p = 0; p < 4; ++p) {
            const int r = arow + 32 * p;
            xv[p] = *(const v4f*)(X + (blockM + r) * K_DIM + kb + acv * 4);
        }
    };
    auto stage_X = [&](int buf) {
#pragma unroll
        for (int p = 0; p < 4; ++p) {
            const int r = arow + 32 * p;
            v4f v = xv[p];
            v4h hi, lo;
            hi.x = (_Float16)v.x; lo.x = (_Float16)(v.x - (float)hi.x);
            hi.y = (_Float16)v.y; lo.y = (_Float16)(v.y - (float)hi.y);
            hi.z = (_Float16)v.z; lo.z = (_Float16)(v.z - (float)hi.z);
            hi.w = (_Float16)v.w; lo.w = (_Float16)(v.w - (float)hi.w);
            *(v4h*)&sAhi[buf][r][acv * 4] = hi;
            *(v4h*)&sAlo[buf][r][acv * 4] = lo;
        }
    };

    // ---- prologue: fill buffer 0 ----
    load_X(0);
    stage_X(0);
    tdm_load_B(0, 0);
    tdm_wait();
    __syncthreads();

    for (int kb = 0; kb < K_DIM; kb += BK) {
        const int  cur  = (kb >> 5) & 1;
        const int  nxt  = cur ^ 1;
        const bool more = (kb + BK) < K_DIM;

        if (more) {
            load_X(kb + BK);          // global loads overlap the WMMA chain below
            tdm_load_B(kb + BK, nxt); // TDM DMA overlaps compute
            if (kb + 2 * BK < K_DIM)  // prefetch two steps ahead
                __builtin_prefetch(X + (blockM + arow) * K_DIM + kb + 2 * BK + acv * 4, 0, 0);
        }

        // ---- compute tile `cur` ----
        v16h bfrag[2];
#pragma unroll
        for (int j = 0; j < 2; ++j) {
            const _Float16* p = &sB[cur][bnr + j * 16][kB];
            v8h b0 = *(const v8h*)p;
            v8h b1 = *(const v8h*)(p + 16);
            bfrag[j] = __builtin_shufflevector(b0, b1,
                0,1,2,3,4,5,6,7,8,9,10,11,12,13,14,15);
        }
#pragma unroll
        for (int i = 0; i < 4; ++i) {
            const _Float16* ph = &sAhi[cur][amr + i * 16][kA];
            v8h h0 = *(const v8h*)ph;
            v8h h1 = *(const v8h*)(ph + 16);
            v16h ahi = __builtin_shufflevector(h0, h1,
                0,1,2,3,4,5,6,7,8,9,10,11,12,13,14,15);
            const _Float16* pl = &sAlo[cur][amr + i * 16][kA];
            v8h l0 = *(const v8h*)pl;
            v8h l1 = *(const v8h*)(pl + 16);
            v16h alo = __builtin_shufflevector(l0, l1,
                0,1,2,3,4,5,6,7,8,9,10,11,12,13,14,15);
#pragma unroll
            for (int j = 0; j < 2; ++j) {
                acc[i][j] = __builtin_amdgcn_wmma_f32_16x16x32_f16(
                    false, ahi, false, bfrag[j], (short)0, acc[i][j], false, false);
                acc[i][j] = __builtin_amdgcn_wmma_f32_16x16x32_f16(
                    false, alo, false, bfrag[j], (short)0, acc[i][j], false, false);
            }
        }

        if (more) {
            stage_X(nxt);   // convert + park X tile k+1 in the other buffer
            tdm_wait();     // wave0: B tile k+1 DMA complete before barrier
        }
        __syncthreads();
    }

    // ---- epilogue: C f32 16x16 layout (lane=N col; VGPR r = row r / r+8) ----
    const int cn    = lane & 15;
    const int rbase = (lane >> 4) * 8;
#pragma unroll
    for (int i = 0; i < 4; ++i)
#pragma unroll
        for (int j = 0; j < 2; ++j) {
            const long row0 = blockM + wm * 64 + i * 16 + rbase;
            const long col  = blockN + wn * 32 + j * 16 + cn;
#pragma unroll
            for (int r = 0; r < 8; ++r)
                C[(row0 + r) * N_DIM + col] = acc[i][j][r] * scale;
        }
}

// ---------------------------------------------------------------------------
extern "C" void kernel_launch(void* const* d_in, const int* in_sizes, int n_in,
                              void* d_out, int out_size, void* d_ws, size_t ws_size,
                              hipStream_t stream) {
    const float* x = (const float*)d_in[0];   // [4,4096,2048] -> [M=16384, K=2048]
    const float* w = (const float*)d_in[1];   // [N=2048, K=2048]
    float* out = (float*)d_out;               // [M, N]

    unsigned*  absbits = (unsigned*)d_ws;                    // 4 bytes
    _Float16*  q       = (_Float16*)((char*)d_ws + 64);      // 2048*2048*2 = 8 MB

    const int nw = in_sizes[1];               // 4194304 weight elements
    const int M  = in_sizes[0] / K_DIM;       // 16384

    hipMemsetAsync(absbits, 0, sizeof(unsigned), stream);
    wabsmax_kernel<<<1024, 256, 0, stream>>>(w, absbits, nw);
    wquant_kernel<<<(nw / 4 + 255) / 256, 256, 0, stream>>>(w, absbits, q, nw);

    dim3 grid(N_DIM / BN, M / BM);            // 16 x 128 blocks
    qlinear_wmma_kernel<<<grid, 256, 0, stream>>>(x, q, absbits, out);
}
